// FlashAttention_81028853006586
// MI455X (gfx1250) — matrix operand order repaired and checked by reading, not compiled, not run on hardware
//
#include <hip/hip_runtime.h>
#include <hip/hip_bf16.h>
#include <stdint.h>

typedef __bf16 bf16_t;
typedef bf16_t v16bf __attribute__((ext_vector_type(16)));
typedef bf16_t v8bf  __attribute__((ext_vector_type(8)));
typedef float  v8f   __attribute__((ext_vector_type(8)));

#define WMMA_BF16(a, b, c) \
  __builtin_amdgcn_wmma_f32_16x16x32_bf16(false, (a), false, (b), (short)0, (c), false, false)

static constexpr int B_ = 2;
static constexpr int T_ = 2048;
static constexpr int C_ = 2048;
static constexpr int H_ = 16;
static constexpr int D_ = 128;      // C/H
static constexpr int M_ = B_ * T_;  // 4096 rows

__device__ __forceinline__ v8f zero_v8f() {
  v8f r;
#pragma unroll
  for (int i = 0; i < 8; ++i) r[i] = 0.0f;
  return r;
}

__device__ __forceinline__ v16bf pack16(v8bf lo, v8bf hi) {
  v16bf f;
#pragma unroll
  for (int e = 0; e < 8; ++e) { f[e] = lo[e]; f[8 + e] = hi[e]; }
  return f;
}

// gfx1250 async copy: 16B global -> LDS, tracked by ASYNCcnt (no VGPR round-trip).
// vdst = VGPR holding LDS byte offset (low 32 bits of generic shared pointer),
// vaddr = 64-bit global address.  (CDNA5 ISA)
__device__ __forceinline__ void async_cp16(const bf16_t* __restrict__ g, bf16_t* l) {
  const uint32_t laddr = (uint32_t)(uintptr_t)l;
  asm volatile("global_load_async_to_lds_b128 %0, %1, off"
               :: "v"(laddr), "v"(g) : "memory");
}
__device__ __forceinline__ void wait_async0() {
  asm volatile("s_wait_asynccnt 0x0" ::: "memory");
}

// ---------------------------------------------------------------- fp32 -> bf16
__global__ void cvt_f32_to_bf16(const float* __restrict__ in, bf16_t* __restrict__ out, int n) {
  int i = blockIdx.x * blockDim.x + threadIdx.x;
  int stride = gridDim.x * blockDim.x;
  for (; i < n; i += stride) out[i] = (bf16_t)in[i];
}

// ------------------------------------------------- C[M,N] = A[M,K] * W[N,K]^T
// 256 threads = 8 waves; block tile 128x128, K-step 32.
// Double-buffered LDS tiles filled with global_load_async_to_lds_b128.
// LDS row stride padded to 48 bf16 (96B = 24 banks) to break b128 conflicts.
static constexpr int KP = 48;  // padded row stride (elements)

template <bool OUT_F32>
__global__ __launch_bounds__(256)
void gemm_bf16_wmma(const bf16_t* __restrict__ A, const bf16_t* __restrict__ W,
                    void* __restrict__ outp, int M, int N, int K) {
  __shared__ bf16_t As[2][128][KP] __attribute__((aligned(32)));
  __shared__ bf16_t Bs[2][128][KP] __attribute__((aligned(32)));

  const int lane = threadIdx.x & 31;
  const int wave = threadIdx.x >> 5;
  const int half = lane >> 4;   // 0: lanes 0-15, 1: lanes 16-31
  const int n16  = lane & 15;
  const int wm = wave & 1;      // 2 waves along M
  const int wn = wave >> 1;     // 4 waves along N
  const int m_blk = blockIdx.y * 128;
  const int n_blk = blockIdx.x * 128;

  // Each thread async-copies 2x16B of A and 2x16B of B per K-step:
  // 128 rows x 4 segments of 16B = 512 chunks, 256 threads.
  auto issue_tile = [&](int buf, int k0) {
#pragma unroll
    for (int i = 0; i < 2; ++i) {
      const int c   = (int)threadIdx.x + i * 256;
      const int row = c >> 2;
      const int seg = c & 3;
      async_cp16(A + (size_t)(m_blk + row) * K + k0 + seg * 8, &As[buf][row][seg * 8]);
      async_cp16(W + (size_t)(n_blk + row) * K + k0 + seg * 8, &Bs[buf][row][seg * 8]);
    }
  };

  v8f acc[4][2];
#pragma unroll
  for (int i = 0; i < 4; ++i)
#pragma unroll
    for (int j = 0; j < 2; ++j) acc[i][j] = zero_v8f();

  const int nsteps = K >> 5;
  issue_tile(0, 0);

  for (int s = 0; s < nsteps; ++s) {
    const int cur = s & 1;
    wait_async0();        // own copies into buf[cur] landed in LDS
    __syncthreads();      // everyone's copies landed; prev buffer free
    if (s + 1 < nsteps) issue_tile(cur ^ 1, (s + 1) << 5);

    // A fragments: lane = row (lane%16); K = half*16 + [0..15] contiguous
    v16bf a[4];
#pragma unroll
    for (int mt = 0; mt < 4; ++mt) {
      const bf16_t* p = &As[cur][wm * 64 + mt * 16 + n16][half * 16];
      a[mt] = pack16(*(const v8bf*)p, *(const v8bf*)(p + 8));
    }
    // B fragments: lane = col (lane%16); K = half*16 + [0..15] contiguous
    v16bf b[2];
#pragma unroll
    for (int nt = 0; nt < 2; ++nt)
      b[nt] = *(const v16bf*)&Bs[cur][wn * 32 + nt * 16 + n16][half * 16];

#pragma unroll
    for (int mt = 0; mt < 4; ++mt)
#pragma unroll
      for (int nt = 0; nt < 2; ++nt)
        acc[mt][nt] = WMMA_BF16(a[mt], b[nt], acc[mt][nt]);
  }

  // C layout: lane holds col n16; VGPR v holds row v + half*8
#pragma unroll
  for (int mt = 0; mt < 4; ++mt)
#pragma unroll
    for (int nt = 0; nt < 2; ++nt)
#pragma unroll
      for (int v = 0; v < 8; ++v) {
        const int row = m_blk + wm * 64 + mt * 16 + v + half * 8;
        const int col = n_blk + wn * 32 + nt * 16 + n16;
        const float val = acc[mt][nt][v];
        if constexpr (OUT_F32)
          ((float*)outp)[(size_t)row * N + col] = val;
        else
          ((bf16_t*)outp)[(size_t)row * N + col] = (bf16_t)val;
      }
}

// ------------------------------------- RoPE(q,k) + head split + V transpose
// qkv: [B*T, 3*C] bf16.  Qb/Kb: [B*H, T, D].  Vt: [B*H, D, T].
__global__ void rope_split_kernel(const bf16_t* __restrict__ qkv,
                                  bf16_t* __restrict__ Qb, bf16_t* __restrict__ Kb,
                                  bf16_t* __restrict__ Vt) {
  const int bid = blockIdx.x;         // b*H*T + h*T + t
  const int t = bid % T_;
  const int h = (bid / T_) % H_;
  const int b = bid / (T_ * H_);
  const bf16_t* qrow = qkv + (size_t)(b * T_ + t) * (3 * C_) + h * D_;
  const bf16_t* krow = qrow + C_;
  const bf16_t* vrow = qrow + 2 * C_;
  const size_t bh = (size_t)(b * H_ + h);
  bf16_t* qo = Qb + (bh * T_ + t) * D_;
  bf16_t* ko = Kb + (bh * T_ + t) * D_;
  const int tid = threadIdx.x;
  if (tid < 64) {
    const int j = tid;
    const float inv_freq = __expf(-(2.0f * j / (float)D_) * 9.210340372f);  // 10000^(-2j/D)
    const float ang = (float)t * inv_freq;
    float s, c;
    __sincosf(ang, &s, &c);
    const float q1 = (float)qrow[j], q2 = (float)qrow[j + 64];
    qo[j]      = (bf16_t)(q1 * c - q2 * s);
    qo[j + 64] = (bf16_t)(q2 * c + q1 * s);
    const float k1 = (float)krow[j], k2 = (float)krow[j + 64];
    ko[j]      = (bf16_t)(k1 * c - k2 * s);
    ko[j + 64] = (bf16_t)(k2 * c + k1 * s);
  }
  for (int d = tid; d < D_; d += blockDim.x)
    Vt[(bh * D_ + d) * T_ + t] = vrow[d];
}

// --------------------------------------------------- causal flash attention
// grid (T/64, H, B), 128 threads = 4 waves; each wave owns 16 query rows.
__global__ __launch_bounds__(128)
void flash_attn_kernel(const bf16_t* __restrict__ Qb, const bf16_t* __restrict__ Kb,
                       const bf16_t* __restrict__ Vt, bf16_t* __restrict__ attnO) {
  __shared__ bf16_t Pbuf[4][16][32] __attribute__((aligned(16)));
  const int lane = threadIdx.x & 31;
  const int wave = threadIdx.x >> 5;
  const int half = lane >> 4;
  const int n16  = lane & 15;
  const int h = blockIdx.y;
  const int b = blockIdx.z;
  const size_t bh = (size_t)(b * H_ + h);
  const int qbase = blockIdx.x * 64 + wave * 16;

  // Q A-fragments across D=128 (4 chunks of 32)
  v16bf aq[4];
  {
    const bf16_t* qrow = Qb + (bh * T_ + qbase + n16) * (size_t)D_;
#pragma unroll
    for (int dc = 0; dc < 4; ++dc) {
      const bf16_t* p = qrow + dc * 32 + half * 16;
      aq[dc] = pack16(*(const v8bf*)p, *(const v8bf*)(p + 8));
    }
  }

  v8f acc[8];
#pragma unroll
  for (int dt = 0; dt < 8; ++dt) acc[dt] = zero_v8f();
  float m_i[8], l_i[8];
#pragma unroll
  for (int v = 0; v < 8; ++v) { m_i[v] = -1e30f; l_i[v] = 0.0f; }

  const float scale = 0.08838834764831845f;  // 1/sqrt(128)
  const int cmax = (qbase + 15) >> 5;        // last 32-key chunk (inclusive)

  for (int c = 0; c <= cmax; ++c) {
    // S = Q * K^T for 32 keys: two 16x16 N-tiles, 4 WMMAs each over D
    v8f st[2];
    st[0] = zero_v8f();
    st[1] = zero_v8f();
#pragma unroll
    for (int tt = 0; tt < 2; ++tt) {
      const int key = c * 32 + tt * 16 + n16;
      const bf16_t* krow = Kb + (bh * T_ + key) * (size_t)D_ + half * 16;
#pragma unroll
      for (int dc = 0; dc < 4; ++dc) {
        const v16bf bk = *(const v16bf*)(krow + dc * 32);
        st[tt] = WMMA_BF16(aq[dc], bk, st[tt]);
      }
    }

    // online softmax (row = v + half*8; reduce across the 16 lanes of each half)
    float p0[8], p1[8], sf[8];
#pragma unroll
    for (int v = 0; v < 8; ++v) {
      const int qr = qbase + v + half * 8;
      float s0 = st[0][v] * scale;
      float s1 = st[1][v] * scale;
      if (c * 32 + n16 > qr)      s0 = -1e30f;   // causal mask
      if (c * 32 + 16 + n16 > qr) s1 = -1e30f;
      float mx = fmaxf(s0, s1);
#pragma unroll
      for (int o = 1; o < 16; o <<= 1) mx = fmaxf(mx, __shfl_xor(mx, o, 32));
      const float m_new = fmaxf(m_i[v], mx);
      sf[v] = __expf(m_i[v] - m_new);
      m_i[v] = m_new;
      p0[v] = __expf(s0 - m_new);
      p1[v] = __expf(s1 - m_new);
      float rs = p0[v] + p1[v];
#pragma unroll
      for (int o = 1; o < 16; o <<= 1) rs += __shfl_xor(rs, o, 32);
      l_i[v] = l_i[v] * sf[v] + rs;
    }
#pragma unroll
    for (int dt = 0; dt < 8; ++dt)
#pragma unroll
      for (int v = 0; v < 8; ++v) acc[dt][v] *= sf[v];

    // P: C-layout -> row-major LDS -> A-fragment (per-wave tile, same-wave RAW)
#pragma unroll
    for (int v = 0; v < 8; ++v) {
      Pbuf[wave][v + half * 8][n16]      = (bf16_t)p0[v];
      Pbuf[wave][v + half * 8][n16 + 16] = (bf16_t)p1[v];
    }
    asm volatile("s_wait_dscnt 0x0" ::: "memory");
    v16bf ap;
    {
      const bf16_t* p = &Pbuf[wave][n16][half * 16];
      ap = pack16(*(const v8bf*)p, *(const v8bf*)(p + 8));
    }
    asm volatile("" ::: "memory");

    // O += P * V  (V pre-transposed: B-fragment = 32B contiguous along T)
    const bf16_t* vbase = Vt + bh * (size_t)D_ * T_ + (size_t)n16 * T_ + c * 32 + half * 16;
#pragma unroll
    for (int dt = 0; dt < 8; ++dt) {
      const v16bf bv = *(const v16bf*)(vbase + (size_t)dt * 16 * T_);
      acc[dt] = WMMA_BF16(ap, bv, acc[dt]);
    }
  }

  // epilogue: O /= l, write bf16 [B*T, C] at column h*128 + dt*16 + n16
#pragma unroll
  for (int v = 0; v < 8; ++v) {
    const float inv_l = 1.0f / l_i[v];
    const int qr = qbase + v + half * 8;
    bf16_t* orow = attnO + (size_t)(b * T_ + qr) * C_ + h * D_ + n16;
#pragma unroll
    for (int dt = 0; dt < 8; ++dt)
      orow[dt * 16] = (bf16_t)(acc[dt][v] * inv_l);
  }
}

// ---------------------------------------------------------------------------
extern "C" void kernel_launch(void* const* d_in, const int* in_sizes, int n_in,
                              void* d_out, int out_size, void* d_ws, size_t ws_size,
                              hipStream_t stream) {
  (void)in_sizes; (void)n_in; (void)out_size; (void)ws_size;
  const float* x    = (const float*)d_in[0];   // [B,T,C]
  const float* wqkv = (const float*)d_in[1];   // [3C,C]
  const float* wo   = (const float*)d_in[2];   // [C,C]
  float* out = (float*)d_out;                  // [B,T,C] fp32

  char* ws = (char*)d_ws;
  size_t off = 0;
  bf16_t* xb    = (bf16_t*)(ws + off); off += (size_t)M_ * C_ * 2;            // 16.8 MB
  bf16_t* wqkvb = (bf16_t*)(ws + off); off += (size_t)3 * C_ * C_ * 2;        // 25.2 MB
  bf16_t* wob   = (bf16_t*)(ws + off); off += (size_t)C_ * C_ * 2;            //  8.4 MB
  bf16_t* qkvb  = (bf16_t*)(ws + off); off += (size_t)M_ * 3 * C_ * 2;        // 50.3 MB
  bf16_t* Qb    = (bf16_t*)(ws + off); off += (size_t)B_ * H_ * T_ * D_ * 2;  // 16.8 MB
  bf16_t* Kb    = (bf16_t*)(ws + off); off += (size_t)B_ * H_ * T_ * D_ * 2;  // 16.8 MB
  bf16_t* Vt    = (bf16_t*)(ws + off); off += (size_t)B_ * H_ * T_ * D_ * 2;  // 16.8 MB
  bf16_t* attnb = (bf16_t*)(ws + off); off += (size_t)M_ * C_ * 2;            // 16.8 MB

  // 1) fp32 -> bf16
  cvt_f32_to_bf16<<<4096, 256, 0, stream>>>(x, xb, M_ * C_);
  cvt_f32_to_bf16<<<4096, 256, 0, stream>>>(wqkv, wqkvb, 3 * C_ * C_);
  cvt_f32_to_bf16<<<4096, 256, 0, stream>>>(wo, wob, C_ * C_);

  // 2) qkv = x * wqkv^T   (M=4096, N=6144, K=2048) -> bf16
  dim3 g1((3 * C_) / 128, M_ / 128);
  gemm_bf16_wmma<false><<<g1, 256, 0, stream>>>(xb, wqkvb, qkvb, M_, 3 * C_, C_);

  // 3) RoPE + split heads + transpose V
  rope_split_kernel<<<B_ * H_ * T_, 128, 0, stream>>>(qkvb, Qb, Kb, Vt);

  // 4) causal flash attention
  dim3 gf(T_ / 64, H_, B_);
  flash_attn_kernel<<<gf, 128, 0, stream>>>(Qb, Kb, Vt, attnb);

  // 5) out = attnO * wo^T  (M=4096, N=2048, K=2048) -> fp32 d_out
  dim3 g2(C_ / 128, M_ / 128);
  gemm_bf16_wmma<true><<<g2, 256, 0, stream>>>(attnb, wob, out, M_, C_, C_);
}
